// ModelSpo_66786741453110
// MI455X (gfx1250) — matrix operand order repaired and checked by reading, not compiled, not run on hardware
//
#include <hip/hip_runtime.h>
#include <hip/hip_bf16.h>
#include <stdint.h>

// Problem constants (from reference): B=64, L=512, D2=512, C=50
#define BSZ   64
#define LSZ   512
#define DSZ   512
#define CSZ   50
#define NPAD  128      // 2 heads x 64 padded columns
#define KTOT  512
#define MROWS 64       // rows per workgroup in main kernel

typedef __attribute__((ext_vector_type(16))) __bf16 v16bf;
typedef __attribute__((ext_vector_type(8)))  float  v8f;

struct U32x8 { uint4 lo, hi; };   // 32 bytes == v16bf

__device__ __forceinline__ unsigned short f2bf_rne(float x) {
    unsigned u = __builtin_bit_cast(unsigned, x);
    u += 0x7FFFu + ((u >> 16) & 1u);
    return (unsigned short)(u >> 16);
}
__device__ __forceinline__ float bf2f(unsigned short h) {
    unsigned u = ((unsigned)h) << 16;
    return __builtin_bit_cast(float, u);
}

// ---------------------------------------------------------------------------
// K1: build column-major padded bf16 hi/lo copies of W[:D2] for both heads,
//     and zero the scalar accumulators. Whi/Wlo layout: [n=0..127][k=0..511].
//     n<64 -> head "start", col c=n; n>=64 -> head "end", col c=n-64; c>=50 pad 0.
// ---------------------------------------------------------------------------
__global__ void prep_w_kernel(const float* __restrict__ Wst,
                              const float* __restrict__ Wen,
                              unsigned short* __restrict__ Whi,
                              unsigned short* __restrict__ Wlo,
                              float* __restrict__ accum) {
    int n = blockIdx.x;              // 0..127
    int h = n >> 6;
    int c = n & 63;
    const float* W = h ? Wen : Wst;
    for (int k = threadIdx.x; k < KTOT; k += blockDim.x) {
        float x = (c < CSZ) ? W[k * CSZ + c] : 0.0f;
        unsigned short hi = f2bf_rne(x);
        float res = x - bf2f(hi);
        unsigned short lo = f2bf_rne(res);
        Whi[(size_t)n * KTOT + k] = hi;
        Wlo[(size_t)n * KTOT + k] = lo;
    }
    if (blockIdx.x == 0 && threadIdx.x == 0) { accum[0] = 0.0f; accum[1] = 0.0f; }
}

// ---------------------------------------------------------------------------
// K2: value_num = sum(text_mask)
// ---------------------------------------------------------------------------
__global__ void mask_sum_kernel(const unsigned char* __restrict__ mask,
                                float* __restrict__ accum) {
    __shared__ float red[256];
    int idx = blockIdx.x * 256 + threadIdx.x;
    red[threadIdx.x] = mask[idx] ? 1.0f : 0.0f;
    __syncthreads();
    for (int s = 128; s > 0; s >>= 1) {
        if (threadIdx.x < s) red[threadIdx.x] += red[threadIdx.x + s];
        __syncthreads();
    }
    if (threadIdx.x == 0) atomicAdd(&accum[0], red[0]);
}

// ---------------------------------------------------------------------------
// K3: sbj_vec[b][d] = mean over l in [start,end] of text_vec[b][l][d]
// ---------------------------------------------------------------------------
__global__ void sbj_vec_kernel(const float* __restrict__ text,
                               const int* __restrict__ sbj,
                               float* __restrict__ sbjv) {
    int b = blockIdx.x;
    int s = sbj[b * 2 + 0];
    int e = sbj[b * 2 + 1];
    float inv = 1.0f / (float)(e - s + 1);
    for (int d = threadIdx.x; d < DSZ; d += blockDim.x) {
        float acc = 0.0f;
        const float* base = text + ((size_t)b * LSZ) * DSZ + d;
        for (int l = s; l <= e; ++l) acc += base[(size_t)l * DSZ];
        sbjv[b * DSZ + d] = acc * inv;
    }
}

// ---------------------------------------------------------------------------
// K4: sbjlog[b][c2] = sbj_vec[b] . W[D2:][:,c] + bias   (c2 = h*50 + c)
// ---------------------------------------------------------------------------
__global__ void sbj_logits_kernel(const float* __restrict__ sbjv,
                                  const float* __restrict__ Wst,
                                  const float* __restrict__ Wen,
                                  const float* __restrict__ bst,
                                  const float* __restrict__ ben,
                                  float* __restrict__ sbjlog) {
    int b = blockIdx.x;
    int c2 = threadIdx.x;
    if (c2 >= 2 * CSZ) return;
    int h = (c2 >= CSZ) ? 1 : 0;
    int c = h ? (c2 - CSZ) : c2;
    const float* W = h ? Wen : Wst;
    float acc = h ? ben[c] : bst[c];
    const float* sv = sbjv + b * DSZ;
    for (int k = 0; k < DSZ; ++k) acc += sv[k] * W[(size_t)(DSZ + k) * CSZ + c];
    sbjlog[b * (2 * CSZ) + c2] = acc;
}

// ---------------------------------------------------------------------------
// K5: main fused GEMM (split-bf16 WMMA) + log-softmax cross-entropy.
//     Block: 64 rows x 128 padded cols; 256 threads = 8 waves, one 16-col
//     WMMA tile per wave, 4 row-tiles each. 3 bf16 WMMAs emulate fp32.
// ---------------------------------------------------------------------------
__global__ __launch_bounds__(256) void main_gemm_ce_kernel(
    const float* __restrict__ text,
    const unsigned short* __restrict__ Whi,
    const unsigned short* __restrict__ Wlo,
    const float* __restrict__ sbjlog,
    const unsigned char* __restrict__ mask,
    const int* __restrict__ objs,
    const int* __restrict__ obje,
    float* __restrict__ accum)
{
    __shared__ unsigned short sAhi[MROWS][32];   // 4 KB
    __shared__ unsigned short sAlo[MROWS][32];   // 4 KB
    __shared__ float sLog[MROWS][NPAD];          // 32 KB
    __shared__ float sRed[256];                  // 1 KB

    const int tid  = threadIdx.x;
    const int lane = tid & 31;
    const int wave = tid >> 5;        // 0..7
    const int n0   = wave * 16;       // this wave's column tile
    const int m0   = blockIdx.x * MROWS;

    v8f acc[4];
    for (int i = 0; i < 4; ++i)
        for (int j = 0; j < 8; ++j) acc[i][j] = 0.0f;

    const int kb  = (lane >> 4) * 8;   // A frag: K runs [kb,kb+8) and [kb+16,kb+24)
    const int kb2 = (lane >> 4) * 16;  // B frag: K runs [kb,kb+8) and [kb+16,kb+24) (same as A)
    const int col = n0 + (lane & 15);  // B frag column (0..127)

    for (int ks = 0; ks < KTOT / 32; ++ks) {
        // ---- stage A (64 rows x 32 K) fp32 -> bf16 hi/lo in LDS ----
        for (int i = 0; i < 2; ++i) {
            int f  = tid + i * 256;        // float4 index 0..511
            int r  = f >> 3;
            int c4 = f & 7;
            const float4 v = *(const float4*)(text + (size_t)(m0 + r) * DSZ + ks * 32 + c4 * 4);
            unsigned short h0 = f2bf_rne(v.x), l0 = f2bf_rne(v.x - bf2f(h0));
            unsigned short h1 = f2bf_rne(v.y), l1 = f2bf_rne(v.y - bf2f(h1));
            unsigned short h2 = f2bf_rne(v.z), l2 = f2bf_rne(v.z - bf2f(h2));
            unsigned short h3 = f2bf_rne(v.w), l3 = f2bf_rne(v.w - bf2f(h3));
            uint2 hp, lp;
            hp.x = (unsigned)h0 | ((unsigned)h1 << 16);
            hp.y = (unsigned)h2 | ((unsigned)h3 << 16);
            lp.x = (unsigned)l0 | ((unsigned)l1 << 16);
            lp.y = (unsigned)l2 | ((unsigned)l3 << 16);
            *(uint2*)&sAhi[r][c4 * 4] = hp;
            *(uint2*)&sAlo[r][c4 * 4] = lp;
        }
        __syncthreads();

        // ---- B fragments straight from L2-resident packed W ----
        const unsigned short* bhp = Whi + (size_t)col * KTOT + ks * 32 + kb;
        const unsigned short* blp = Wlo + (size_t)col * KTOT + ks * 32 + kb;
        U32x8 qbh, qbl;
        qbh.lo = *(const uint4*)bhp;      qbh.hi = *(const uint4*)(bhp + 16);
        qbl.lo = *(const uint4*)blp;      qbl.hi = *(const uint4*)(blp + 16);
        v16bf bh = __builtin_bit_cast(v16bf, qbh);
        v16bf bl = __builtin_bit_cast(v16bf, qbl);

        for (int rt = 0; rt < 4; ++rt) {
            int row = rt * 16 + (lane & 15);
            U32x8 qah, qal;
            qah.lo = *(const uint4*)&sAhi[row][kb];
            qah.hi = *(const uint4*)&sAhi[row][kb + 16];
            qal.lo = *(const uint4*)&sAlo[row][kb];
            qal.hi = *(const uint4*)&sAlo[row][kb + 16];
            v16bf ah = __builtin_bit_cast(v16bf, qah);
            v16bf al = __builtin_bit_cast(v16bf, qal);
            // fp32-emulating split: Ahi*Bhi + Alo*Bhi + Ahi*Blo
            acc[rt] = __builtin_amdgcn_wmma_f32_16x16x32_bf16(
                false, ah, false, bh, (short)0, acc[rt], false, false);
            acc[rt] = __builtin_amdgcn_wmma_f32_16x16x32_bf16(
                false, al, false, bh, (short)0, acc[rt], false, false);
            acc[rt] = __builtin_amdgcn_wmma_f32_16x16x32_bf16(
                false, ah, false, bl, (short)0, acc[rt], false, false);
        }
        __syncthreads();
    }

    // ---- scatter accumulators to LDS logits (C/D layout: VGPR r -> row r(+8)) ----
    {
        int rowoff = (lane >> 4) * 8;
        int nc = n0 + (lane & 15);
        for (int rt = 0; rt < 4; ++rt)
            for (int r = 0; r < 8; ++r)
                sLog[rt * 16 + rowoff + r][nc] = acc[rt][r];
    }
    __syncthreads();

    // ---- fused masked cross-entropy: one thread per (row, head) ----
    float contrib = 0.0f;
    if (tid < MROWS * 2) {
        int r = tid >> 1;
        int h = tid & 1;
        int g = m0 + r;                 // global row = b*L + l
        int b = g >> 9;
        const float* sadd = sbjlog + b * (2 * CSZ) + h * CSZ;
        const float* lrow = &sLog[r][h * 64];
        float mx = -3.4e38f;
        for (int c = 0; c < CSZ; ++c) {
            float v = lrow[c] + sadd[c];
            mx = fmaxf(mx, v);
        }
        float se = 0.0f;
        for (int c = 0; c < CSZ; ++c)
            se += __expf(lrow[c] + sadd[c] - mx);
        int lab = (h ? obje : objs)[g];
        float lv = lrow[lab] + sadd[lab];
        float nll = (mx + __logf(se)) - lv;
        contrib = mask[g] ? nll : 0.0f;
    }
    sRed[tid] = contrib;
    __syncthreads();
    for (int s = 128; s > 0; s >>= 1) {
        if (tid < s) sRed[tid] += sRed[tid + s];
        __syncthreads();
    }
    if (tid == 0) atomicAdd(&accum[1], sRed[0]);
}

// ---------------------------------------------------------------------------
// K6: loss = nll_sum / value_num
// ---------------------------------------------------------------------------
__global__ void finalize_kernel(const float* __restrict__ accum,
                                float* __restrict__ out) {
    out[0] = accum[1] / accum[0];
}

extern "C" void kernel_launch(void* const* d_in, const int* in_sizes, int n_in,
                              void* d_out, int out_size, void* d_ws, size_t ws_size,
                              hipStream_t stream) {
    (void)in_sizes; (void)n_in; (void)out_size; (void)ws_size;
    const float*         text  = (const float*)d_in[0];
    const unsigned char* maskp = (const unsigned char*)d_in[1];   // bool
    const int*           sbj   = (const int*)d_in[2];
    const int*           objs  = (const int*)d_in[3];
    const int*           obje  = (const int*)d_in[4];
    const float*         Wst   = (const float*)d_in[5];
    const float*         bst   = (const float*)d_in[6];
    const float*         Wen   = (const float*)d_in[7];
    const float*         ben   = (const float*)d_in[8];
    float* out = (float*)d_out;

    // workspace layout
    char* ws = (char*)d_ws;
    unsigned short* Whi = (unsigned short*)(ws);                       // 128*512*2 = 128 KB
    unsigned short* Wlo = (unsigned short*)(ws + 131072);              // 128 KB
    float* fbase  = (float*)(ws + 262144);
    float* accum  = fbase;            // [0]=value_num, [1]=nll sum
    float* sbjv   = fbase + 8;        // 64*512 floats
    float* sbjlog = sbjv + BSZ * DSZ; // 64*100 floats

    prep_w_kernel   <<<NPAD, 256, 0, stream>>>(Wst, Wen, Whi, Wlo, accum);
    mask_sum_kernel <<<(BSZ * LSZ) / 256, 256, 0, stream>>>(maskp, accum);
    sbj_vec_kernel  <<<BSZ, 256, 0, stream>>>(text, sbj, sbjv);
    sbj_logits_kernel<<<BSZ, 128, 0, stream>>>(sbjv, Wst, Wen, bst, ben, sbjlog);
    main_gemm_ce_kernel<<<(BSZ * LSZ) / MROWS, 256, 0, stream>>>(
        text, Whi, Wlo, sbjlog, maskp, objs, obje, accum);
    finalize_kernel <<<1, 1, 0, stream>>>(accum, out);
}
